// TransformerBlock_37125697307259
// MI455X (gfx1250) — matrix operand order repaired and checked by reading, not compiled, not run on hardware
//
#include <hip/hip_runtime.h>

typedef unsigned short u16;
typedef __bf16 bf16_t;
typedef __attribute__((ext_vector_type(16))) __bf16 v16bf;
typedef __attribute__((ext_vector_type(8)))  __bf16 v8bf;
typedef __attribute__((ext_vector_type(8)))  float  v8f;
typedef __attribute__((ext_vector_type(4)))  unsigned int v4u;
typedef __attribute__((ext_vector_type(8)))  int v8i;
typedef __attribute__((ext_vector_type(4)))  int v4i;

#define B_  4
#define S_  2048
#define D_  512
#define H_  8
#define KD_ 64
#define FF_ 2048
#define HKD_ (H_*KD_)   // 512

__device__ __forceinline__ bf16_t f2bf(float f) {
  unsigned u = __builtin_bit_cast(unsigned, f);
  u += 0x7FFFu + ((u >> 16) & 1u);              // round-to-nearest-even
  u16 h = (u16)(u >> 16);
  return __builtin_bit_cast(bf16_t, h);
}

__device__ __forceinline__ v16bf cat8(v8bf lo, v8bf hi) {
  return __builtin_shufflevector(lo, hi, 0,1,2,3,4,5,6,7,8,9,10,11,12,13,14,15);
}

// ---------------------------------------------------------------- conversion
__global__ void cvt_f32_bf16(const float* __restrict__ in, bf16_t* __restrict__ out, int n) {
  int i = blockIdx.x * blockDim.x + threadIdx.x;
  if (i < n) out[i] = f2bf(in[i]);
}

// ---------------------------------------------------------------- V transpose
// vb: [B,S,H,KD] bf16  ->  vt: [B,H,KD,S] bf16
__global__ void transpose_v(const bf16_t* __restrict__ vb, bf16_t* __restrict__ vt, int n) {
  int i = blockIdx.x * blockDim.x + threadIdx.x;
  if (i >= n) return;
  int s = i & (S_ - 1);
  int f = (i >> 11) & (KD_ - 1);
  int h = (i >> 17) & (H_ - 1);
  int b = i >> 20;
  vt[i] = vb[(((size_t)(b * S_ + s)) * H_ + h) * KD_ + f];
}

// ---------------------------------------------------------------- bf16 GEMM
// C[M,N](f32 accum) = A[M,K]bf16 @ B[K,N]bf16 + bias ; optional ReLU
// Block: 256 thr (8 waves). Block tile 128x128, K step 32. Wave tile 32x64.
// A tile is DMA'd into LDS by the Tensor Data Mover (tensor_load_to_lds,
// TENSORcnt); B tile is staged transposed by the vector pipe.
template<bool RELU>
__global__ __launch_bounds__(256)
void gemm_bf16(const bf16_t* __restrict__ A, const bf16_t* __restrict__ Bm,
               const float* __restrict__ bias,
               float* __restrict__ outF, bf16_t* __restrict__ outH,
               int M, int N, int K) {
  __shared__ __attribute__((aligned(32))) bf16_t lA[128 * 32];   // row-major [128][32]
  __shared__ __attribute__((aligned(32))) bf16_t lBt[128 * 32];  // transposed [N=128][K=32]
  const int tid  = threadIdx.x;
  const int lane = tid & 31, wv = tid >> 5;
  const int half = lane >> 4, ln = lane & 15;
  const int wm = wv & 3, wn = wv >> 2;
  const int mbase = blockIdx.y * 128;
  const int nbase = blockIdx.x * 128;

  // ---- TDM descriptor, invariant parts (uniform -> SGPRs)
  const unsigned ldsA = (unsigned)(unsigned long long)(void*)&lA[0];
  v8i g1;
  g1[0] = 0x00010000;                               // data_size = 2B; mask=0
  g1[1] = (K & 0xFFFF) << 16;                       // tensor_dim0[15:0] (atomic addr 0)
  g1[2] = ((unsigned)K >> 16) | ((M & 0xFFFF) << 16);  // tensor_dim0 hi | tensor_dim1 lo
  g1[3] = (((unsigned)M >> 16) & 0xFFFF) | (32u << 16); // tensor_dim1 hi | tile_dim0=32
  g1[4] = 128;                                      // tile_dim1=128, tile_dim2=0
  g1[5] = K;                                        // tensor_dim0_stride (data_size units)
  g1[6] = 0;
  g1[7] = 0;
  const v4i z4 = {0, 0, 0, 0};
  const v8i z8 = {0, 0, 0, 0, 0, 0, 0, 0};
  const unsigned long long gbaseA =
      (unsigned long long)(const void*)A + (size_t)mbase * K * 2;

  const v8f vzero = {0,0,0,0,0,0,0,0};
  v8f acc[2][4];
  #pragma unroll
  for (int i = 0; i < 2; i++)
    #pragma unroll
    for (int j = 0; j < 4; j++) acc[i][j] = vzero;

  for (int kb = 0; kb < K; kb += 32) {
    // ---- A tile via Tensor Data Mover (issued once, EXEC-ignored)
    if (tid == 0) {
      unsigned long long ga = gbaseA + (unsigned long long)kb * 2;
      v4u g0;
      g0[0] = 1u;                                   // count=1 (valid user D#)
      g0[1] = ldsA;                                 // lds_addr
      g0[2] = (unsigned)(ga & 0xFFFFFFFFu);         // global_addr[31:0]
      g0[3] = (unsigned)((ga >> 32) & 0x1FFFFFFu) | (2u << 30); // addr[56:32] | type=2
      __builtin_amdgcn_tensor_load_to_lds(g0, g1, z4, z4, z8, 0);
    }
    if (kb + 32 < K)  // gfx1250 global_prefetch of next B K-tile
      __builtin_prefetch(Bm + (size_t)(kb + 32 + (tid >> 3)) * N + nbase, 0, 1);
    // ---- stage B tile transposed
    #pragma unroll
    for (int i = 0; i < 2; i++) {
      int c = tid + 256 * i;
      int k = c >> 4, ng = c & 15;
      v8bf bv = *(const v8bf*)(Bm + (size_t)(kb + k) * N + nbase + ng * 8);
      #pragma unroll
      for (int e = 0; e < 8; e++) lBt[(ng * 8 + e) * 32 + k] = bv[e];
    }
    if (tid == 0) __builtin_amdgcn_s_wait_tensorcnt(0);
    __syncthreads();

    v16bf af[2];
    #pragma unroll
    for (int mt = 0; mt < 2; mt++) {
      const bf16_t* ap = &lA[(wm * 32 + mt * 16 + ln) * 32];
      af[mt] = *(const v16bf*)(ap + half * 16);
    }
    v16bf bfg[4];
    #pragma unroll
    for (int nt = 0; nt < 4; nt++)
      bfg[nt] = *(const v16bf*)&lBt[(wn * 64 + nt * 16 + ln) * 32 + half * 16];

    #pragma unroll
    for (int mt = 0; mt < 2; mt++)
      #pragma unroll
      for (int nt = 0; nt < 4; nt++)
        acc[mt][nt] = __builtin_amdgcn_wmma_f32_16x16x32_bf16(
            false, af[mt], false, bfg[nt], (short)0, acc[mt][nt], false, false);
    __syncthreads();
  }

  #pragma unroll
  for (int mt = 0; mt < 2; mt++) {
    #pragma unroll
    for (int nt = 0; nt < 4; nt++) {
      int col = nbase + wn * 64 + nt * 16 + ln;
      float bs = bias ? bias[col] : 0.0f;
      #pragma unroll
      for (int r = 0; r < 8; r++) {
        int row = mbase + wm * 32 + mt * 16 + r + 8 * half;
        float v = acc[mt][nt][r] + bs;
        if (RELU) v = fmaxf(v, 0.0f);
        if (outF) outF[(size_t)row * N + col] = v;
        if (outH) outH[(size_t)row * N + col] = f2bf(v);
      }
    }
  }
}

// ---------------------------------------------------------------- attention
// qb,kb: [B,S,H,KD] bf16 ; vt: [B,H,KD,S] bf16 ; out ctxb: [B,S,H,KD] bf16
// One wave = one 16-query tile, flash-style online softmax over keys (chunks of 32).
__global__ __launch_bounds__(128)
void attention(const bf16_t* __restrict__ qb, const bf16_t* __restrict__ kb,
               const bf16_t* __restrict__ vt, bf16_t* __restrict__ ctxb) {
  __shared__ __attribute__((aligned(32))) bf16_t P[4][16][32];
  const int lane = threadIdx.x & 31;
  const int wv   = threadIdx.x >> 5;
  const int half = lane >> 4, ln = lane & 15;
  const int wg = blockIdx.x * 4 + wv;
  const int qt = wg & 127;            // S/16 = 128
  const int h  = (wg >> 7) & 7;
  const int b  = wg >> 10;            // H*(S/16) = 1024
  const int s0 = qt * 16;

  // Q fragments (constant over key loop), KD=64 -> two K chunks of 32
  const bf16_t* qrow = qb + (size_t)(b * S_ + s0 + ln) * HKD_ + h * KD_;
  v16bf qa[2];
  #pragma unroll
  for (int kc = 0; kc < 2; kc++)
    qa[kc] = *(const v16bf*)(qrow + kc * 32 + half * 16);

  const v8f vzero = {0,0,0,0,0,0,0,0};
  float m[8], l[8];
  v8f ctx[4];
  #pragma unroll
  for (int r = 0; r < 8; r++) { m[r] = -1e30f; l[r] = 0.0f; }
  #pragma unroll
  for (int t = 0; t < 4; t++) ctx[t] = vzero;

  const bf16_t* kbase = kb + (size_t)b * S_ * HKD_ + h * KD_;
  const bf16_t* vbase = vt + (size_t)(b * H_ + h) * KD_ * S_;

  for (int k0 = 0; k0 < S_; k0 += 32) {
    // ---- scores: two 16x16 tiles over this 32-key chunk
    v8f sc[2]; sc[0] = vzero; sc[1] = vzero;
    #pragma unroll
    for (int nt = 0; nt < 2; nt++)
      #pragma unroll
      for (int kc = 0; kc < 2; kc++) {
        v16bf kf = *(const v16bf*)(kbase + (size_t)(k0 + nt * 16 + ln) * HKD_
                                   + kc * 32 + half * 16);
        sc[nt] = __builtin_amdgcn_wmma_f32_16x16x32_bf16(
            false, qa[kc], false, kf, (short)0, sc[nt], false, false);
      }
    // ---- online softmax update (rows live in 16-lane groups)
    #pragma unroll
    for (int r = 0; r < 8; r++) {
      float a0 = sc[0][r] * 0.125f;           // 1/sqrt(KD)
      float a1 = sc[1][r] * 0.125f;
      float bm = fmaxf(a0, a1);
      bm = fmaxf(bm, __shfl_xor(bm, 1));
      bm = fmaxf(bm, __shfl_xor(bm, 2));
      bm = fmaxf(bm, __shfl_xor(bm, 4));
      bm = fmaxf(bm, __shfl_xor(bm, 8));
      float mn   = fmaxf(m[r], bm);
      float corr = __expf(m[r] - mn);
      float p0   = __expf(a0 - mn);
      float p1   = __expf(a1 - mn);
      float rs   = p0 + p1;
      rs += __shfl_xor(rs, 1);
      rs += __shfl_xor(rs, 2);
      rs += __shfl_xor(rs, 4);
      rs += __shfl_xor(rs, 8);
      l[r] = l[r] * corr + rs;
      m[r] = mn;
      #pragma unroll
      for (int t = 0; t < 4; t++) ctx[t][r] *= corr;
      P[wv][r + 8 * half][ln]      = f2bf(p0);
      P[wv][r + 8 * half][16 + ln] = f2bf(p1);
    }
    // ---- P (C-layout) -> A-layout fragment via LDS
    const bf16_t* pr = &P[wv][ln][0];
    v16bf pa = *(const v16bf*)(pr + half * 16);
    // ---- ctx += P @ V  (4 feature tiles of 16)
    #pragma unroll
    for (int t = 0; t < 4; t++) {
      v16bf vf = *(const v16bf*)(vbase + (size_t)(t * 16 + ln) * S_ + k0 + half * 16);
      ctx[t] = __builtin_amdgcn_wmma_f32_16x16x32_bf16(
          false, pa, false, vf, (short)0, ctx[t], false, false);
    }
  }
  // ---- normalize and store bf16 ctx
  #pragma unroll
  for (int t = 0; t < 4; t++) {
    #pragma unroll
    for (int r = 0; r < 8; r++) {
      float v = ctx[t][r] / l[r];
      ctxb[(size_t)(b * S_ + s0 + r + 8 * half) * HKD_ + h * KD_ + t * 16 + ln] = f2bf(v);
    }
  }
}

// ---------------------------------------------------------------- add + LayerNorm
// out = LN(a + b) * g + be ; row length D=512 ; one block per row
__global__ __launch_bounds__(256)
void add_ln(const float* __restrict__ a, const float* __restrict__ bsrc,
            const float* __restrict__ g, const float* __restrict__ be,
            float* __restrict__ outF, bf16_t* __restrict__ outH) {
  __shared__ float ws1[8], ws2[8];
  const int row = blockIdx.x;
  const int tid = threadIdx.x;
  const float* ap = a + (size_t)row * D_;
  const float* bp = bsrc + (size_t)row * D_;
  float x0 = ap[tid] + bp[tid];
  float x1 = ap[tid + 256] + bp[tid + 256];
  float s1 = x0 + x1, s2 = x0 * x0 + x1 * x1;
  for (int msk = 1; msk < 32; msk <<= 1) {
    s1 += __shfl_xor(s1, msk);
    s2 += __shfl_xor(s2, msk);
  }
  if ((tid & 31) == 0) { ws1[tid >> 5] = s1; ws2[tid >> 5] = s2; }
  __syncthreads();
  float t1 = 0.f, t2 = 0.f;
  #pragma unroll
  for (int i = 0; i < 8; i++) { t1 += ws1[i]; t2 += ws2[i]; }
  float mean = t1 * (1.0f / D_);
  float var  = t2 * (1.0f / D_) - mean * mean;
  float rstd = rsqrtf(var + 1e-6f);
  float y0 = (x0 - mean) * rstd * g[tid] + be[tid];
  float y1 = (x1 - mean) * rstd * g[tid + 256] + be[tid + 256];
  size_t o = (size_t)row * D_;
  if (outF) { outF[o + tid] = y0; outF[o + tid + 256] = y1; }
  if (outH) { outH[o + tid] = f2bf(y0); outH[o + tid + 256] = f2bf(y1); }
}

// ---------------------------------------------------------------- launch
extern "C" void kernel_launch(void* const* d_in, const int* in_sizes, int n_in,
                              void* d_out, int out_size, void* d_ws, size_t ws_size,
                              hipStream_t stream) {
  (void)in_sizes; (void)n_in; (void)out_size; (void)ws_size;
  const float* x   = (const float*)d_in[0];
  const float* wq  = (const float*)d_in[1];
  const float* bq  = (const float*)d_in[2];
  const float* wk  = (const float*)d_in[3];
  const float* bk  = (const float*)d_in[4];
  const float* wvp = (const float*)d_in[5];
  const float* bv  = (const float*)d_in[6];
  const float* wo  = (const float*)d_in[7];
  const float* bo  = (const float*)d_in[8];
  const float* w1  = (const float*)d_in[9];
  const float* b1  = (const float*)d_in[10];
  const float* w2  = (const float*)d_in[11];
  const float* b2  = (const float*)d_in[12];
  const float* g1  = (const float*)d_in[13];
  const float* be1 = (const float*)d_in[14];
  const float* g2  = (const float*)d_in[15];
  const float* be2 = (const float*)d_in[16];
  float* out = (float*)d_out;

  char* w = (char*)d_ws;
  size_t off = 0;
  auto take = [&](size_t bytes) -> char* {
    char* p = w + off;
    off += (bytes + 255) & ~(size_t)255;
    return p;
  };
  const size_t NTOK = (size_t)B_ * S_;                    // 8192 rows
  bf16_t* xb    = (bf16_t*)take(NTOK * D_ * 2);
  bf16_t* wqb   = (bf16_t*)take((size_t)D_ * HKD_ * 2);
  bf16_t* wkb   = (bf16_t*)take((size_t)D_ * HKD_ * 2);
  bf16_t* wvb   = (bf16_t*)take((size_t)D_ * HKD_ * 2);
  bf16_t* wob   = (bf16_t*)take((size_t)HKD_ * D_ * 2);
  bf16_t* w1b   = (bf16_t*)take((size_t)D_ * FF_ * 2);
  bf16_t* w2b   = (bf16_t*)take((size_t)FF_ * D_ * 2);
  bf16_t* qbuf  = (bf16_t*)take(NTOK * HKD_ * 2);
  bf16_t* kbuf  = (bf16_t*)take(NTOK * HKD_ * 2);
  bf16_t* vbuf  = (bf16_t*)take(NTOK * HKD_ * 2);
  bf16_t* vtb   = (bf16_t*)take(NTOK * HKD_ * 2);
  bf16_t* ctxb  = (bf16_t*)take(NTOK * HKD_ * 2);
  float*  attno = (float*) take(NTOK * D_ * 4);
  float*  out1  = (float*) take(NTOK * D_ * 4);
  bf16_t* out1b = (bf16_t*)take(NTOK * D_ * 2);
  bf16_t* hidb  = (bf16_t*)take(NTOK * FF_ * 2);
  float*  ffn   = (float*) take(NTOK * D_ * 4);

  auto cvt = [&](const float* src, bf16_t* dst, int n) {
    cvt_f32_bf16<<<(n + 255) / 256, 256, 0, stream>>>(src, dst, n);
  };
  cvt(x,   xb,  (int)(NTOK * D_));
  cvt(wq,  wqb, D_ * HKD_);
  cvt(wk,  wkb, D_ * HKD_);
  cvt(wvp, wvb, D_ * HKD_);
  cvt(wo,  wob, HKD_ * D_);
  cvt(w1,  w1b, D_ * FF_);
  cvt(w2,  w2b, FF_ * D_);

  // QKV projections: [8192,512] @ [512,512] -> bf16
  dim3 gqkv(HKD_ / 128, (unsigned)(NTOK / 128));
  gemm_bf16<false><<<gqkv, 256, 0, stream>>>(xb, wqb, bq, nullptr, qbuf, (int)NTOK, HKD_, D_);
  gemm_bf16<false><<<gqkv, 256, 0, stream>>>(xb, wkb, bk, nullptr, kbuf, (int)NTOK, HKD_, D_);
  gemm_bf16<false><<<gqkv, 256, 0, stream>>>(xb, wvb, bv, nullptr, vbuf, (int)NTOK, HKD_, D_);

  // V transpose to [B,H,KD,S]
  {
    int n = (int)(NTOK * HKD_);
    transpose_v<<<(n + 255) / 256, 256, 0, stream>>>(vbuf, vtb, n);
  }

  // flash attention: 4096 query tiles / 4 waves per block
  attention<<<(B_ * H_ * (S_ / 16)) / 4, 128, 0, stream>>>(qbuf, kbuf, vtb, ctxb);

  // output projection -> fp32
  dim3 go(D_ / 128, (unsigned)(NTOK / 128));
  gemm_bf16<false><<<go, 256, 0, stream>>>(ctxb, wob, bo, attno, nullptr, (int)NTOK, D_, HKD_);

  // out1 = LN(x + attn_out)
  add_ln<<<(unsigned)NTOK, 256, 0, stream>>>(x, attno, g1, be1, out1, out1b);

  // FFN
  dim3 g1d(FF_ / 128, (unsigned)(NTOK / 128));
  gemm_bf16<true><<<g1d, 256, 0, stream>>>(out1b, w1b, b1, nullptr, hidb, (int)NTOK, FF_, D_);
  dim3 g2d(D_ / 128, (unsigned)(NTOK / 128));
  gemm_bf16<false><<<g2d, 256, 0, stream>>>(hidb, w2b, b2, ffn, nullptr, (int)NTOK, D_, FF_);

  // final LN -> d_out (fp32)
  add_ln<<<(unsigned)NTOK, 256, 0, stream>>>(out1, ffn, g2, be2, out, nullptr);
}
